// DiT_8529805050403
// MI455X (gfx1250) — matrix operand order repaired and checked by reading, not compiled, not run on hardware
//
#include <hip/hip_runtime.h>
#include <hip/hip_bf16.h>

typedef __attribute__((ext_vector_type(16))) __bf16 v16bf;
typedef __attribute__((ext_vector_type(8)))  __bf16 v8bf;
typedef __attribute__((ext_vector_type(8)))  float  v8f;

constexpr int kHS  = 1024;
constexpr int kTOK = 256;   // tokens per image
constexpr int kB   = 32;
constexpr int kNL  = 12;

// ---------------- CDNA5 async global->LDS copy (ASYNCcnt-tracked) ----------------
// Each active lane copies 16B from its global address to its LDS address.
__device__ __forceinline__ void async_copy_b128(const __bf16* lds_dst, const void* gsrc) {
  unsigned laddr = (unsigned)(size_t)lds_dst;           // low 32 bits = LDS offset
  unsigned long long gaddr = (unsigned long long)(size_t)gsrc;
  asm volatile("global_load_async_to_lds_b128 %0, %1, off"
               :: "v"(laddr), "v"(gaddr) : "memory");
}
__device__ __forceinline__ void wait_async() {
  asm volatile("s_wait_asynccnt 0x0" ::: "memory");
}

// ---------------- WMMA fragment loaders (bf16, from LDS) ----------------
// A-operand layout (16x32): lane l -> row (l&15); elems 0..7 -> k = base+i,
// elems 8..15 -> k = 16+base+i, base = (l>>4)*8. Two contiguous 16B reads.
__device__ __forceinline__ v16bf ld_frag_a(const __bf16* base, int stride, int lane) {
  const __bf16* p = base + (lane & 15) * stride + ((lane >> 4) << 3);
  v8bf lo = *(const v8bf*)(p);
  v8bf hi = *(const v8bf*)(p + 16);
  v16bf r;
#pragma unroll
  for (int i = 0; i < 8; ++i) { r[i] = lo[i]; r[i + 8] = hi[i]; }
  return r;
}
// B-operand layout (from N x K rows): lane l -> column n=(l&15);
// elems 0..7 -> k = base+i, elems 8..15 -> k = 16+base+i, base = (l>>4)*8.
__device__ __forceinline__ v16bf ld_frag_b(const __bf16* base, int stride, int lane) {
  const __bf16* p = base + (lane & 15) * stride + ((lane >> 4) << 3);
  v8bf lo = *(const v8bf*)(p);
  v8bf hi = *(const v8bf*)(p + 16);
  v16bf r;
#pragma unroll
  for (int i = 0; i < 8; ++i) { r[i] = lo[i]; r[i + 8] = hi[i]; }
  return r;
}

__device__ __forceinline__ v8f wmma_bf16(v16bf a, v16bf b, v8f c) {
  return __builtin_amdgcn_wmma_f32_16x16x32_bf16(false, a, false, b, (short)0, c, false, false);
}

__device__ __forceinline__ unsigned pack_bf16(float a, float b) {
  union { __bf16 h[2]; unsigned u; } p;
  p.h[0] = (__bf16)a; p.h[1] = (__bf16)b;
  return p.u;
}

// ---------------- Generic GEMM: C[M,N] = epi(A_bf16[M,K] * W_f32[K,N] + bias) ----
// EPI: 0 f32 out; 1 f32 out + pos_embed[(m%256),n]; 2 bf16 out * scale;
//      3 bf16 gelu; 4 f32 X[m,n] += gate[(m/256)*auxStride+n]*v; 5 unpatchify to d_out
template <int EPI>
__global__ __launch_bounds__(256) void gemm_bf16(
    const __bf16* __restrict__ A, const float* __restrict__ W,
    const float* __restrict__ bias, void* __restrict__ Out,
    const float* __restrict__ aux, int auxStride,
    int M, int N, int K, float scale) {
  __shared__ __bf16 sA[128 * 40];
  __shared__ __bf16 sB[128 * 40];
  const int tid  = threadIdx.x;
  const int lane = tid & 31;
  const int wid  = tid >> 5;
  const int wm   = wid >> 2;  // 0..1
  const int wn   = wid & 3;   // 0..3
  const int m0   = blockIdx.y * 128;
  const int n0   = blockIdx.x * 128;

  v8f acc[4][2];
#pragma unroll
  for (int mi = 0; mi < 4; ++mi)
#pragma unroll
    for (int ni = 0; ni < 2; ++ni)
#pragma unroll
      for (int e = 0; e < 8; ++e) acc[mi][ni][e] = 0.0f;

  for (int k0 = 0; k0 < K; k0 += 32) {
    // --- issue async copies for the A tile (128 x 32 bf16) first ---
#pragma unroll
    for (int i = 0; i < 2; ++i) {
      int s = tid + i * 256;
      int r = s >> 2, c8 = (s & 3) << 3;
      int gm = m0 + r;
      if (gm < M) {
        async_copy_b128(&sA[r * 40 + c8], A + (size_t)gm * K + k0 + c8);
      } else {
        v8bf z;
#pragma unroll
        for (int j = 0; j < 8; ++j) z[j] = (__bf16)0.0f;
        *(v8bf*)(&sA[r * 40 + c8]) = z;
      }
    }
    // --- stage W tile transposed while the async copies are in flight ---
    // Two consecutive k-rows per thread; pack bf16 pairs and store dwords.
#pragma unroll
    for (int i = 0; i < 2; ++i) {
      int u = tid + i * 256;         // 512 units: 16 k-pairs x 32 n-quads
      int kk = (u >> 5) << 1;        // even k
      int nq = (u & 31) << 2;
      const float4 w0 = *(const float4*)(W + (size_t)(k0 + kk) * N + n0 + nq);
      const float4 w1 = *(const float4*)(W + (size_t)(k0 + kk + 1) * N + n0 + nq);
      *(unsigned*)(&sB[(nq + 0) * 40 + kk]) = pack_bf16(w0.x, w1.x);
      *(unsigned*)(&sB[(nq + 1) * 40 + kk]) = pack_bf16(w0.y, w1.y);
      *(unsigned*)(&sB[(nq + 2) * 40 + kk]) = pack_bf16(w0.z, w1.z);
      *(unsigned*)(&sB[(nq + 3) * 40 + kk]) = pack_bf16(w0.w, w1.w);
    }
    wait_async();
    __syncthreads();
    v16bf bf[2];
#pragma unroll
    for (int ni = 0; ni < 2; ++ni)
      bf[ni] = ld_frag_b(&sB[(wn * 32 + ni * 16) * 40], 40, lane);
#pragma unroll
    for (int mi = 0; mi < 4; ++mi) {
      v16bf af = ld_frag_a(&sA[(wm * 64 + mi * 16) * 40], 40, lane);
#pragma unroll
      for (int ni = 0; ni < 2; ++ni) acc[mi][ni] = wmma_bf16(af, bf[ni], acc[mi][ni]);
    }
    __syncthreads();
  }

  const int ncol  = lane & 15;
  const int rbase = (lane >> 4) << 3;
#pragma unroll
  for (int mi = 0; mi < 4; ++mi)
#pragma unroll
    for (int ni = 0; ni < 2; ++ni) {
      int n = n0 + wn * 32 + ni * 16 + ncol;
      float bv = bias[n];
#pragma unroll
      for (int e = 0; e < 8; ++e) {
        int m = m0 + wm * 64 + mi * 16 + rbase + e;
        if (m >= M) continue;
        float v = acc[mi][ni][e] + bv;
        if (EPI == 0) {
          ((float*)Out)[(size_t)m * N + n] = v;
        } else if (EPI == 1) {
          ((float*)Out)[(size_t)m * N + n] = v + aux[(size_t)(m & 255) * N + n];
        } else if (EPI == 2) {
          ((__bf16*)Out)[(size_t)m * N + n] = (__bf16)(v * scale);
        } else if (EPI == 3) {
          float g = 0.5f * v * (1.0f + erff(v * 0.70710678118654752f));
          ((__bf16*)Out)[(size_t)m * N + n] = (__bf16)g;
        } else if (EPI == 4) {
          float g = aux[(size_t)(m >> 8) * auxStride + n];
          ((float*)Out)[(size_t)m * N + n] += g * v;
        } else if (EPI == 5) {
          // unpatchify: m = b*256 + l, n = ph*128 + pw*64 + d
          int b = m >> 8, l = m & 255;
          int hp = l >> 4, wp = l & 15;
          int ph = n >> 7, pw = (n >> 6) & 1, d = n & 63;
          size_t oi = (((size_t)(b * 64 + d) * 32) + (size_t)(hp * 2 + ph)) * 32 + (size_t)(wp * 2 + pw);
          ((float*)Out)[oi] = v;
        }
      }
    }
}

// ---------------- patchify: x(B,64,32,32) -> bf16 A-matrix (8192 x 256) ---------
__global__ __launch_bounds__(256) void patchify_k(const float* __restrict__ x,
                                                  __bf16* __restrict__ tokA) {
  int idx = blockIdx.x * 256 + threadIdx.x;  // 32*256*256
  int b = idx >> 16;
  int r = idx & 65535;
  int l = r >> 8, f = r & 255;
  int hp = l >> 4, wp = l & 15;
  int ph = f >> 7, pw = (f >> 6) & 1, d = f & 63;
  float v = x[(((size_t)(b * 64 + d) * 32) + (size_t)(hp * 2 + ph)) * 32 + (size_t)(wp * 2 + pw)];
  tokA[idx] = (__bf16)v;
}

// ---------------- conditioning: c_act = silu([cos,sin](t*freqs) + emb[y]) -------
__global__ __launch_bounds__(256) void cond_k(const int* __restrict__ y,
                                              const int* __restrict__ t,
                                              const float* __restrict__ emb,
                                              __bf16* __restrict__ cact) {
  int b = blockIdx.x;
  float tv = (float)t[b];
  int yb = y[b];
#pragma unroll
  for (int i = 0; i < 4; ++i) {
    int j = threadIdx.x + i * 256;
    int f = (j < 512) ? j : (j - 512);
    float freq = expf(-logf(10000.0f) * (float)f / 512.0f);
    float ang = tv * freq;
    float base = (j < 512) ? cosf(ang) : sinf(ang);
    float c = base + emb[(size_t)yb * kHS + j];
    float ca = c / (1.0f + expf(-c));
    cact[(size_t)b * kHS + j] = (__bf16)ca;
  }
}

// ---------------- LayerNorm + adaLN modulation -> bf16 --------------------------
__global__ __launch_bounds__(256) void ln_mod_k(const float* __restrict__ X,
                                                const float* __restrict__ shift,
                                                const float* __restrict__ scl,
                                                int modStride,
                                                __bf16* __restrict__ Hx) {
  int m = blockIdx.x;
  int b = m >> 8;
  const float* xr = X + (size_t)m * kHS;
  float vals[4];
  float s = 0.0f, s2 = 0.0f;
#pragma unroll
  for (int i = 0; i < 4; ++i) {
    float v = xr[threadIdx.x + i * 256];
    vals[i] = v; s += v; s2 += v * v;
  }
  __shared__ float red[256];
  red[threadIdx.x] = s; __syncthreads();
  for (int off = 128; off > 0; off >>= 1) {
    if (threadIdx.x < off) red[threadIdx.x] += red[threadIdx.x + off];
    __syncthreads();
  }
  float mu = red[0] * (1.0f / kHS);
  __syncthreads();
  red[threadIdx.x] = s2; __syncthreads();
  for (int off = 128; off > 0; off >>= 1) {
    if (threadIdx.x < off) red[threadIdx.x] += red[threadIdx.x + off];
    __syncthreads();
  }
  float var = red[0] * (1.0f / kHS) - mu * mu;
  float rstd = rsqrtf(var + 1e-5f);
#pragma unroll
  for (int i = 0; i < 4; ++i) {
    int c = threadIdx.x + i * 256;
    float sc = scl[(size_t)b * modStride + c];
    float sh = shift[(size_t)b * modStride + c];
    Hx[(size_t)m * kHS + c] = (__bf16)((vals[i] - mu) * rstd * (1.0f + sc) + sh);
  }
}

// ---------------- WMMA attention: one WG per (b, h, 64-query block) -------------
__global__ __launch_bounds__(128) void attn_k(const __bf16* __restrict__ Q,
                                              const __bf16* __restrict__ Kk,
                                              const __bf16* __restrict__ Vv,
                                              __bf16* __restrict__ O) {
  __shared__ __bf16 sQ[64 * 72];
  __shared__ __bf16 sK[64 * 72];
  __shared__ __bf16 sV[64 * 72];  // transposed: [d][key]
  __shared__ __bf16 sP[64 * 72];
  const int tid = threadIdx.x, lane = tid & 31, wid = tid >> 5;
  const int b = blockIdx.z, h = blockIdx.y, qb = blockIdx.x;
  const int bc = h * 64;

  // stage Q block via async copies (queries pre-scaled by 1/8 in the q-GEMM)
#pragma unroll
  for (int i = 0; i < 4; ++i) {
    int s = tid + i * 128;
    int r = s >> 3, c8 = (s & 7) << 3;
    async_copy_b128(&sQ[r * 72 + c8],
                    Q + ((size_t)(b * 256 + qb * 64 + r)) * kHS + bc + c8);
  }
  wait_async();
  __syncthreads();
  v16bf aq[2];
  aq[0] = ld_frag_a(&sQ[(wid * 16) * 72 + 0], 72, lane);
  aq[1] = ld_frag_a(&sQ[(wid * 16) * 72 + 32], 72, lane);

  v8f S[16];
#pragma unroll
  for (int kt = 0; kt < 16; ++kt)
#pragma unroll
    for (int e = 0; e < 8; ++e) S[kt][e] = 0.0f;

  // ---- S = Q * K^T, streaming keys in chunks of 64 ----
  for (int kc = 0; kc < 4; ++kc) {
#pragma unroll
    for (int i = 0; i < 4; ++i) {
      int s = tid + i * 128;
      int r = s >> 3, c8 = (s & 7) << 3;
      async_copy_b128(&sK[r * 72 + c8],
                      Kk + ((size_t)(b * 256 + kc * 64 + r)) * kHS + bc + c8);
    }
    wait_async();
    __syncthreads();
#pragma unroll
    for (int t2 = 0; t2 < 4; ++t2) {
      int kt = kc * 4 + t2;
#pragma unroll
      for (int ds = 0; ds < 2; ++ds) {
        v16bf bk = ld_frag_b(&sK[(t2 * 16) * 72 + ds * 32], 72, lane);
        S[kt] = wmma_bf16(aq[ds], bk, S[kt]);
      }
    }
    __syncthreads();
  }

  // ---- row softmax (each lane holds 8 rows' values for one column) ----
#pragma unroll
  for (int e = 0; e < 8; ++e) {
    float mx = S[0][e];
#pragma unroll
    for (int kt = 1; kt < 16; ++kt) mx = fmaxf(mx, S[kt][e]);
    mx = fmaxf(mx, __shfl_xor(mx, 1, 32));
    mx = fmaxf(mx, __shfl_xor(mx, 2, 32));
    mx = fmaxf(mx, __shfl_xor(mx, 4, 32));
    mx = fmaxf(mx, __shfl_xor(mx, 8, 32));
    float sum = 0.0f;
#pragma unroll
    for (int kt = 0; kt < 16; ++kt) {
      float ev = __expf(S[kt][e] - mx);
      S[kt][e] = ev; sum += ev;
    }
    sum += __shfl_xor(sum, 1, 32);
    sum += __shfl_xor(sum, 2, 32);
    sum += __shfl_xor(sum, 4, 32);
    sum += __shfl_xor(sum, 8, 32);
    float inv = 1.0f / sum;
#pragma unroll
    for (int kt = 0; kt < 16; ++kt) S[kt][e] *= inv;
  }

  // ---- O = P * V, streaming keys in chunks of 64 ----
  v8f Oa[4];
#pragma unroll
  for (int n = 0; n < 4; ++n)
#pragma unroll
    for (int e = 0; e < 8; ++e) Oa[n][e] = 0.0f;

  const int rb0 = (lane >> 4) << 3;
  const int nc0 = lane & 15;
  for (int kc = 0; kc < 4; ++kc) {
    // write this wave's P chunk (rows wid*16..+16, cols 0..63) as bf16
#pragma unroll
    for (int t2 = 0; t2 < 4; ++t2) {
      int kt = kc * 4 + t2;
      int col = t2 * 16 + nc0;
#pragma unroll
      for (int e = 0; e < 8; ++e)
        sP[(wid * 16 + rb0 + e) * 72 + col] = (__bf16)S[kt][e];
    }
    // stage V chunk transposed: sV[d][key] (needs register transpose)
#pragma unroll
    for (int i = 0; i < 4; ++i) {
      int s = tid + i * 128;
      int kk = s >> 3, d8 = (s & 7) << 3;
      v8bf val = *(const v8bf*)(Vv + ((size_t)(b * 256 + kc * 64 + kk)) * kHS + bc + d8);
#pragma unroll
      for (int j = 0; j < 8; ++j) sV[(d8 + j) * 72 + kk] = val[j];
    }
    __syncthreads();
    v16bf ap[2];
    ap[0] = ld_frag_a(&sP[(wid * 16) * 72 + 0], 72, lane);
    ap[1] = ld_frag_a(&sP[(wid * 16) * 72 + 32], 72, lane);
#pragma unroll
    for (int n = 0; n < 4; ++n)
#pragma unroll
      for (int ks = 0; ks < 2; ++ks) {
        v16bf bv = ld_frag_b(&sV[(n * 16) * 72 + ks * 32], 72, lane);
        Oa[n] = wmma_bf16(ap[ks], bv, Oa[n]);
      }
    __syncthreads();
  }
#pragma unroll
  for (int n = 0; n < 4; ++n)
#pragma unroll
    for (int e = 0; e < 8; ++e)
      O[((size_t)(b * 256 + qb * 64 + wid * 16 + rb0 + e)) * kHS + bc + n * 16 + nc0] =
          (__bf16)Oa[n][e];
}

// ================================================================================
extern "C" void kernel_launch(void* const* d_in, const int* in_sizes, int n_in,
                              void* d_out, int out_size, void* d_ws, size_t ws_size,
                              hipStream_t stream) {
  (void)in_sizes; (void)n_in; (void)out_size; (void)ws_size;
  const float* x        = (const float*)d_in[0];
  const int*   y        = (const int*)d_in[1];
  const int*   t        = (const int*)d_in[2];
  const float* proj_w   = (const float*)d_in[3];
  const float* proj_b   = (const float*)d_in[4];
  const float* pos      = (const float*)d_in[5];
  const float* emb      = (const float*)d_in[6];
  const float* mod_w    = (const float*)d_in[7];
  const float* mod_b    = (const float*)d_in[8];
  const float* wq       = (const float*)d_in[9];
  const float* bq       = (const float*)d_in[10];
  const float* wk       = (const float*)d_in[11];
  const float* bk       = (const float*)d_in[12];
  const float* wv       = (const float*)d_in[13];
  const float* bv       = (const float*)d_in[14];
  const float* wo       = (const float*)d_in[15];
  const float* bo       = (const float*)d_in[16];
  const float* f1w      = (const float*)d_in[17];
  const float* f1b      = (const float*)d_in[18];
  const float* f2w      = (const float*)d_in[19];
  const float* f2b      = (const float*)d_in[20];
  const float* fmw      = (const float*)d_in[21];
  const float* fmb      = (const float*)d_in[22];
  const float* fow      = (const float*)d_in[23];
  const float* fob      = (const float*)d_in[24];
  float* out = (float*)d_out;

  char* ws = (char*)d_ws;
  size_t off = 0;
  auto alloc = [&](size_t bytes) -> void* {
    void* p = ws + off;
    off += (bytes + 255) & ~(size_t)255;
    return p;
  };
  const size_t MTOK = (size_t)kB * kTOK;  // 8192
  float*  X    = (float*)alloc(MTOK * kHS * 4);
  __bf16* hx   = (__bf16*)alloc(MTOK * kHS * 2);
  __bf16* qbuf = (__bf16*)alloc(MTOK * kHS * 2);
  __bf16* kbuf = (__bf16*)alloc(MTOK * kHS * 2);
  __bf16* vbuf = (__bf16*)alloc(MTOK * kHS * 2);
  __bf16* obuf = (__bf16*)alloc(MTOK * kHS * 2);
  __bf16* h1   = (__bf16*)alloc(MTOK * 4096 * 2);
  __bf16* tokA = (__bf16*)alloc(MTOK * 256 * 2);
  __bf16* cact = (__bf16*)alloc((size_t)kB * kHS * 2);
  float*  mod  = (float*)alloc((size_t)kB * 6144 * 4);
  float*  fmod = (float*)alloc((size_t)kB * 2048 * 4);

  dim3 blk(256);
  dim3 gTok(kHS / 128, (int)(MTOK / 128));   // 8 x 64

  cond_k<<<kB, blk, 0, stream>>>(y, t, emb, cact);
  patchify_k<<<(int)(MTOK * 256 / 256), blk, 0, stream>>>(x, tokA);
  // tok = patchify @ proj_w + proj_b + pos_embed
  gemm_bf16<1><<<gTok, blk, 0, stream>>>(tokA, proj_w, proj_b, X, pos, 0,
                                         (int)MTOK, kHS, 256, 1.0f);

  for (int l = 0; l < kNL; ++l) {
    const float* mwL = mod_w + (size_t)l * kHS * 6144;
    const float* mbL = mod_b + (size_t)l * 6144;
    // mod = c_act @ mw + mb   (M=32, N=6144, K=1024)
    gemm_bf16<0><<<dim3(48, 1), blk, 0, stream>>>(cact, mwL, mbL, mod, nullptr, 0,
                                                  kB, 6144, kHS, 1.0f);
    // hx = ln(X)*(1+sc_a)+sh_a
    ln_mod_k<<<(int)MTOK, blk, 0, stream>>>(X, mod + 0 * kHS, mod + 1 * kHS, 6144, hx);
    // q,k,v  (q pre-scaled by 1/sqrt(dh)=0.125)
    gemm_bf16<2><<<gTok, blk, 0, stream>>>(hx, wq + (size_t)l * kHS * kHS,
                                           bq + (size_t)l * kHS, qbuf, nullptr, 0,
                                           (int)MTOK, kHS, kHS, 0.125f);
    gemm_bf16<2><<<gTok, blk, 0, stream>>>(hx, wk + (size_t)l * kHS * kHS,
                                           bk + (size_t)l * kHS, kbuf, nullptr, 0,
                                           (int)MTOK, kHS, kHS, 1.0f);
    gemm_bf16<2><<<gTok, blk, 0, stream>>>(hx, wv + (size_t)l * kHS * kHS,
                                           bv + (size_t)l * kHS, vbuf, nullptr, 0,
                                           (int)MTOK, kHS, kHS, 1.0f);
    // attention
    attn_k<<<dim3(4, 16, 32), dim3(128), 0, stream>>>(qbuf, kbuf, vbuf, obuf);
    // X += g_a * (o @ wo + bo)
    gemm_bf16<4><<<gTok, blk, 0, stream>>>(obuf, wo + (size_t)l * kHS * kHS,
                                           bo + (size_t)l * kHS, X, mod + 2 * kHS, 6144,
                                           (int)MTOK, kHS, kHS, 1.0f);
    // hx = ln(X)*(1+sc_m)+sh_m
    ln_mod_k<<<(int)MTOK, blk, 0, stream>>>(X, mod + 3 * kHS, mod + 4 * kHS, 6144, hx);
    // h1 = gelu(hx @ f1 + b1)
    gemm_bf16<3><<<dim3(32, (int)(MTOK / 128)), blk, 0, stream>>>(
        hx, f1w + (size_t)l * kHS * 4096, f1b + (size_t)l * 4096, h1, nullptr, 0,
        (int)MTOK, 4096, kHS, 1.0f);
    // X += g_m * (h1 @ f2 + b2)
    gemm_bf16<4><<<gTok, blk, 0, stream>>>(h1, f2w + (size_t)l * 4096 * kHS,
                                           f2b + (size_t)l * kHS, X, mod + 5 * kHS, 6144,
                                           (int)MTOK, kHS, 4096, 1.0f);
  }

  // final modulation + projection + unpatchify
  gemm_bf16<0><<<dim3(16, 1), blk, 0, stream>>>(cact, fmw, fmb, fmod, nullptr, 0,
                                                kB, 2048, kHS, 1.0f);
  ln_mod_k<<<(int)MTOK, blk, 0, stream>>>(X, fmod + 0 * kHS, fmod + 1 * kHS, 2048, hx);
  gemm_bf16<5><<<dim3(2, (int)(MTOK / 128)), blk, 0, stream>>>(
      hx, fow, fob, out, nullptr, 0, (int)MTOK, 256, kHS, 1.0f);
}
